// DecoderRNN_11510512353619
// MI455X (gfx1250) — matrix operand order repaired and checked by reading, not compiled, not run on hardware
//
#include <hip/hip_runtime.h>

// ---------------------------------------------------------------------------
// CDNA5 (gfx1250) attention-LSTM decoder.
// Big GEMMs run on v_wmma_f32_16x16x32_f16 (fp16 in / fp32 accumulate);
// weights cast to fp16 once per launch (resident in 192MB L2 across steps).
// ---------------------------------------------------------------------------

typedef __attribute__((ext_vector_type(16))) _Float16 v16h;
typedef __attribute__((ext_vector_type(8)))  _Float16 v8h;
typedef __attribute__((ext_vector_type(8)))  float    v8f;

#define BATCH 32
#define HDIM  1024
#define EDIM  512
#define VDIM  128
#define KQDIM 128
#define SDIM  256
#define VOCAB 8000
#define K1    1664   // E + V + H   (LSTM1 fused input)
#define K2    2048   // H + H       (LSTM2 fused input)
#define KO    1152   // H + V       (output projection input)
#define NG    4096   // 4*H gates

static __device__ __forceinline__ float sigm(float x) { return 1.0f / (1.0f + expf(-x)); }

// A-fragment (16x32 f16): two contiguous 8-half chunks at +0/+16 (K layout per ISA 7.12.2)
static __device__ __forceinline__ v16h ld_a(const _Float16* p) {
  v8h lo = *(const v8h*)(p);
  v8h hi = *(const v8h*)(p + 16);
  v16h r;
#pragma unroll
  for (int i = 0; i < 8; ++i) { r[i] = lo[i]; r[i + 8] = hi[i]; }
  return r;
}

// out[m][n0+n] = sum_k A[m][k] * W[n0+n][k] + bias[n0+n]
// A: [32, K] row-major fp16; W: [N, K] row-major fp16; out row stride = ldo.
__global__ void __launch_bounds__(256) gemm_wmma_kernel(
    const _Float16* __restrict__ A,
    const _Float16* __restrict__ W,
    const float*    __restrict__ bias,
    float*          __restrict__ out,
    int K, int N, long ldo)
{
  const int wave = blockIdx.x * (blockDim.x >> 5) + (threadIdx.x >> 5);
  const int n0 = wave << 4;
  if (n0 >= N) return;                      // wave-uniform: EXEC stays all-ones
  const int lane = threadIdx.x & 31;
  const int r    = lane & 15;
  const int hi   = lane >> 4;
  const _Float16* pa0 = A + (long)r        * K + (hi << 3);   // A rows 0..15
  const _Float16* pa1 = A + (long)(r + 16) * K + (hi << 3);   // A rows 16..31
  const _Float16* pb  = W + (long)(n0 + r) * K + (hi << 3);   // B col strip (same K layout as A)
  v8f acc0 = {};
  v8f acc1 = {};
  for (int kc = 0; kc < K; kc += 32) {
    v16h a0 = ld_a(pa0 + kc);
    v16h a1 = ld_a(pa1 + kc);
    v16h b  = ld_a(pb + kc);
    acc0 = __builtin_amdgcn_wmma_f32_16x16x32_f16(false, a0, false, b, (short)0, acc0, false, false);
    acc1 = __builtin_amdgcn_wmma_f32_16x16x32_f16(false, a1, false, b, (short)0, acc1, false, false);
  }
  const int   n  = n0 + r;
  const float bv = bias ? bias[n] : 0.0f;
  const int   mb = hi << 3;
#pragma unroll
  for (int i = 0; i < 8; ++i) {
    out[(long)(mb + i)      * ldo + n] = acc0[i] + bv;
    out[(long)(mb + i + 16) * ldo + n] = acc1[i] + bv;
  }
}

// LSTM cell pointwise: gates [32,4096] (i,f,g,o) -> c (in/out), h -> two fp16 dests (+opt fp32)
__global__ void __launch_bounds__(256) cell_kernel(
    const float* __restrict__ gates,
    float*       __restrict__ c,
    float*       __restrict__ hf32,           // may be null
    _Float16*    __restrict__ hd1, int ld1, int off1,
    _Float16*    __restrict__ hd2, int ld2, int off2)
{
  const int idx = blockIdx.x * blockDim.x + threadIdx.x;   // 32*1024 threads
  const int b = idx >> 10, j = idx & 1023;
  const float* g = gates + (long)b * NG;
  const float gi = g[j], gf = g[j + 1024], gg = g[j + 2048], go = g[j + 3072];
  const float cn = sigm(gf) * c[idx] + sigm(gi) * tanhf(gg);
  c[idx] = cn;
  const float h = sigm(go) * tanhf(cn);
  if (hf32) hf32[idx] = h;
  const _Float16 hh = (_Float16)h;
  hd1[(long)b * ld1 + off1 + j] = hh;
  hd2[(long)b * ld2 + off2 + j] = hh;
}

// Attention (one workgroup per batch row): q = h2@Wq^T+bq ; softmax(keys.q) ; ctx = attn@values.
// Also stages ctx (fp16) into xh1/oin and copies emb_{t+1} into xh1.
__global__ void __launch_bounds__(256) attn_kernel(
    const float* __restrict__ h2f,     // [32,1024]
    const float* __restrict__ Wq,      // [128,1024]
    const float* __restrict__ bq,      // [128]
    const float* __restrict__ keys,    // [32,256,128]
    const float* __restrict__ values,  // [32,256,128]
    _Float16*    __restrict__ xh1,     // [32,K1]
    _Float16*    __restrict__ oin,     // [32,KO]
    const _Float16* __restrict__ emb_all, // [T,32,512]
    int t, int T)
{
  const int b = blockIdx.x, tid = threadIdx.x;
  __shared__ float q[KQDIM];
  __shared__ float attn[SDIM];
  __shared__ float red[256];

  // q (2 threads per output over 512-element halves)
  {
    const int k = tid & 127, half = tid >> 7;
    const float* h = h2f + (long)b * HDIM + half * 512;
    const float* w = Wq  + (long)k * HDIM + half * 512;
    float s = 0.0f;
    for (int i = 0; i < 512; ++i) s += h[i] * w[i];
    red[tid] = s;
  }
  __syncthreads();
  if (tid < KQDIM) q[tid] = red[tid] + red[tid + 128] + bq[tid];
  __syncthreads();

  // energy[s] = keys[b,s,:].q
  float e;
  {
    const float* kr = keys + ((long)b * SDIM + tid) * KQDIM;
    float s = 0.0f;
    for (int i = 0; i < KQDIM; ++i) s += kr[i] * q[i];
    e = s;
  }
  red[tid] = e; __syncthreads();
  for (int off = 128; off > 0; off >>= 1) {
    if (tid < off) red[tid] = fmaxf(red[tid], red[tid + off]);
    __syncthreads();
  }
  const float mx = red[0]; __syncthreads();
  const float ex = expf(e - mx);
  red[tid] = ex; __syncthreads();
  for (int off = 128; off > 0; off >>= 1) {
    if (tid < off) red[tid] += red[tid + off];
    __syncthreads();
  }
  const float inv = 1.0f / red[0]; __syncthreads();
  attn[tid] = ex * inv;
  __syncthreads();

  // context[v] = sum_s attn[s] * values[b,s,v]
  if (tid < VDIM) {
    const float* vb = values + (long)b * SDIM * VDIM + tid;
    float ctx = 0.0f;
    for (int s = 0; s < SDIM; ++s) ctx += attn[s] * vb[(long)s * VDIM];
    const _Float16 ch = (_Float16)ctx;
    xh1[(long)b * K1 + EDIM + tid] = ch;     // next step's x = [emb | ctx | h1]
    oin[(long)b * KO + HDIM + tid] = ch;     // this step's  [h2 | ctx]
  }
  // stage emb_{t+1}
  if (t + 1 < T) {
    const _Float16* src = emb_all + ((long)(t + 1) * BATCH + b) * EDIM;
    _Float16* dst = xh1 + (long)b * K1;
    for (int i = tid; i < EDIM; i += 256) dst[i] = src[i];
  }
}

// fp32 [R,C] -> fp16 at dst[r*dstride + doff + c]
__global__ void __launch_bounds__(256) cast_copy_kernel(
    const float* __restrict__ src, _Float16* __restrict__ dst,
    long total, int C, int dstride, int doff)
{
  const long idx = (long)blockIdx.x * blockDim.x + threadIdx.x;
  if (idx >= total) return;
  const long r = idx / C;
  const int  c = (int)(idx - r * C);
  dst[r * dstride + doff + c] = (_Float16)src[idx];
}

// emb_all[t,b,e] = fp16(embedding[decoder_inputs[b,t], e])
__global__ void __launch_bounds__(256) embed_kernel(
    const int* __restrict__ toks,        // [32, T+1]
    const float* __restrict__ emb,       // [VOCAB, 512]
    _Float16* __restrict__ out, int T)
{
  const long idx = (long)blockIdx.x * blockDim.x + threadIdx.x;
  if (idx >= (long)T * BATCH * EDIM) return;
  const int e = (int)(idx & (EDIM - 1));
  const int b = (int)((idx >> 9) & (BATCH - 1));
  const int t = (int)(idx >> 14);
  const int tok = toks[b * (T + 1) + t];
  out[idx] = (_Float16)emb[(long)tok * EDIM + e];
}

__global__ void __launch_bounds__(256) init_kernel(
    const float* __restrict__ enc_h,     // [32,1024]
    const _Float16* __restrict__ emb_all,
    _Float16* __restrict__ xh1, _Float16* __restrict__ xh2,
    float* __restrict__ h2f, float* __restrict__ c1, float* __restrict__ c2)
{
  const int idx = blockIdx.x * blockDim.x + threadIdx.x;  // 32*2048
  const int b = idx >> 11, j = idx & 2047;
  if (j < 1024) {
    const float h = enc_h[b * HDIM + j];
    const _Float16 hh = (_Float16)h;
    xh1[(long)b * K1 + EDIM + VDIM + j] = hh;  // h1 = enc_h
    xh2[(long)b * K2 + j]               = hh;  // h1 slot
    xh2[(long)b * K2 + HDIM + j]        = hh;  // h2 = enc_h
    h2f[b * HDIM + j] = h;
    c1[b * HDIM + j] = 0.0f;
    c2[b * HDIM + j] = 0.0f;
  } else {
    const int jj = j - 1024;
    if (jj < EDIM)             xh1[(long)b * K1 + jj] = emb_all[(long)b * EDIM + jj]; // emb_0
    else if (jj < EDIM + VDIM) xh1[(long)b * K1 + jj] = (_Float16)0.0f;               // ctx_0 = 0
  }
}

static inline int nblk(long n) { return (int)((n + 255) / 256); }

extern "C" void kernel_launch(void* const* d_in, const int* in_sizes, int n_in,
                              void* d_out, int out_size, void* d_ws, size_t ws_size,
                              hipStream_t stream) {
  const int*   toks   = (const int*)  d_in[0];   // [32, T+1]
  const float* enc_h  = (const float*)d_in[2];   // [32,1024]
  const float* keys   = (const float*)d_in[3];   // [32,256,128]
  const float* values = (const float*)d_in[4];   // [32,256,128]
  const float* embed  = (const float*)d_in[5];   // [8000,512]
  const float* W_ih1  = (const float*)d_in[6];   // [4096,640]
  const float* W_hh1  = (const float*)d_in[7];   // [4096,1024]
  const float* b1     = (const float*)d_in[8];
  const float* W_ih2  = (const float*)d_in[9];   // [4096,1024]
  const float* W_hh2  = (const float*)d_in[10];  // [4096,1024]
  const float* b2     = (const float*)d_in[11];
  const float* W_q    = (const float*)d_in[12];  // [128,1024]
  const float* b_q    = (const float*)d_in[13];
  const float* W_out  = (const float*)d_in[14];  // [8000,1152]
  const float* b_out  = (const float*)d_in[15];

  const int T = in_sizes[0] / BATCH - 1;         // 128

  // workspace carve-up (256B aligned)
  char* w = (char*)d_ws;
  size_t off = 0;
  auto carve = [&](size_t bytes) -> void* {
    void* p = w + off;
    off += (bytes + 255) & ~(size_t)255;
    return p;
  };
  _Float16* Wc1  = (_Float16*)carve((size_t)NG * K1 * 2);      // [4096,1664]
  _Float16* Wc2  = (_Float16*)carve((size_t)NG * K2 * 2);      // [4096,2048]
  _Float16* Wo   = (_Float16*)carve((size_t)VOCAB * KO * 2);   // [8000,1152]
  _Float16* emba = (_Float16*)carve((size_t)T * BATCH * EDIM * 2);
  _Float16* xh1  = (_Float16*)carve((size_t)BATCH * K1 * 2);
  _Float16* xh2  = (_Float16*)carve((size_t)BATCH * K2 * 2);
  _Float16* oin  = (_Float16*)carve((size_t)BATCH * KO * 2);
  float* g1  = (float*)carve((size_t)BATCH * NG * 4);
  float* g2  = (float*)carve((size_t)BATCH * NG * 4);
  float* c1  = (float*)carve((size_t)BATCH * HDIM * 4);
  float* c2  = (float*)carve((size_t)BATCH * HDIM * 4);
  float* h2f = (float*)carve((size_t)BATCH * HDIM * 4);
  (void)ws_size; (void)n_in; (void)out_size;

  // --- prep: fp16 weight concat, embedding gather, state init ---
  cast_copy_kernel<<<nblk((long)NG * 640),     256, 0, stream>>>(W_ih1, Wc1, (long)NG * 640,  640,  K1, 0);
  cast_copy_kernel<<<nblk((long)NG * HDIM),    256, 0, stream>>>(W_hh1, Wc1, (long)NG * HDIM, HDIM, K1, 640);
  cast_copy_kernel<<<nblk((long)NG * HDIM),    256, 0, stream>>>(W_ih2, Wc2, (long)NG * HDIM, HDIM, K2, 0);
  cast_copy_kernel<<<nblk((long)NG * HDIM),    256, 0, stream>>>(W_hh2, Wc2, (long)NG * HDIM, HDIM, K2, HDIM);
  cast_copy_kernel<<<nblk((long)VOCAB * KO),   256, 0, stream>>>(W_out, Wo,  (long)VOCAB * KO, KO,  KO, 0);
  embed_kernel<<<nblk((long)T * BATCH * EDIM), 256, 0, stream>>>(toks, embed, emba, T);
  init_kernel<<<nblk((long)BATCH * 2048),      256, 0, stream>>>(enc_h, emba, xh1, xh2, h2f, c1, c2);

  const int gblk_gates = (NG / 16 + 7) / 8;       // 32 blocks (256 waves)
  const int gblk_out   = (VOCAB / 16 + 7) / 8;    // 63 blocks (500 live waves)
  const long ldo_out   = (long)T * VOCAB;         // d_out is [B, T, VOCAB]

  for (int t = 0; t < T; ++t) {
    // LSTM1: gates1 = [emb|ctx|h1] @ Wc1^T + b1
    gemm_wmma_kernel<<<gblk_gates, 256, 0, stream>>>(xh1, Wc1, b1, g1, K1, NG, NG);
    cell_kernel<<<128, 256, 0, stream>>>(g1, c1, nullptr,
                                         xh2, K2, 0,            // h1 -> LSTM2 input
                                         xh1, K1, EDIM + VDIM); // h1 -> next step x
    // LSTM2: gates2 = [h1|h2] @ Wc2^T + b2
    gemm_wmma_kernel<<<gblk_gates, 256, 0, stream>>>(xh2, Wc2, b2, g2, K2, NG, NG);
    cell_kernel<<<128, 256, 0, stream>>>(g2, c2, h2f,
                                         xh2, K2, HDIM,         // h2 -> LSTM2 hidden
                                         oin, KO, 0);           // h2 -> output proj input
    // attention: writes ctx into xh1/oin, stages emb_{t+1}
    attn_kernel<<<BATCH, 256, 0, stream>>>(h2f, W_q, b_q, keys, values,
                                           xh1, oin, emba, t, T);
    // logits_t = [h2|ctx] @ W_out^T + b_out  -> d_out[:, t, :]
    gemm_wmma_kernel<<<gblk_out, 256, 0, stream>>>(oin, Wo, b_out,
                                                   (float*)d_out + (long)t * VOCAB,
                                                   KO, VOCAB, ldo_out);
  }
}
